// MultiheadSelfAttention_8735963480198
// MI455X (gfx1250) — matrix operand order repaired and checked by reading, not compiled, not run on hardware
//
#include <hip/hip_runtime.h>
#include <hip/hip_bf16.h>
#include <cstdint>

#define B_  2
#define T_  4096
#define M_  1024
#define H_  8
#define D_  128
#define BT_ (B_ * T_)      // 8192
#define HD_ (H_ * D_)      // 1024

typedef __attribute__((ext_vector_type(16))) __bf16 bf16x16;
typedef __attribute__((ext_vector_type(8)))  float  f32x8;

union Frag { bf16x16 v; uint32_t u[8]; uint4 q[2]; };
union Acc  { f32x8   v; float    f[8]; };

__device__ __forceinline__ uint16_t f2bf(float f) {
    uint32_t u = __float_as_uint(f);
    return (uint16_t)((u + 0x7FFFu + ((u >> 16) & 1u)) >> 16);
}

__device__ __forceinline__ f32x8 wmma_bf16(bf16x16 a, bf16x16 b, f32x8 c) {
    return __builtin_amdgcn_wmma_f32_16x16x32_bf16(false, a, false, b,
                                                   (short)0, c, false, false);
}

// A-frag: base points at (row start + chunk) elem 0; hi = lane>=16.
// Per-lane dwords: elems {0..7}+8*hi and {16..23}+8*hi -> b128 at hi*16, hi*16+32.
__device__ __forceinline__ Frag load_afrag(const char* base, int hi) {
    Frag f;
    f.q[0] = *(const uint4*)(base + hi * 16);
    f.q[1] = *(const uint4*)(base + hi * 16 + 32);
    return f;
}
// B-frag: per-lane elems {0..7}+8*hi and {16..23}+8*hi -> b128 at hi*16, hi*16+32.
__device__ __forceinline__ Frag load_bfrag(const char* base, int hi) {
    Frag f;
    f.q[0] = *(const uint4*)(base + hi * 16);
    f.q[1] = *(const uint4*)(base + hi * 16 + 32);
    return f;
}

// ---------------------------------------------------------------------------
// x f32 -> bf16, 4 elems/thread
__global__ void k_conv_x4(const float4* __restrict__ in,
                          ushort4* __restrict__ out, int n4) {
    int i = blockIdx.x * blockDim.x + threadIdx.x;
    if (i >= n4) return;
    float4 v = in[i];
    ushort4 o;
    o.x = f2bf(v.x); o.y = f2bf(v.y); o.z = f2bf(v.z); o.w = f2bf(v.w);
    out[i] = o;
}

// Weight f32 [K x N] -> bf16 transposed [N x K]  (K = N = 1024)
__global__ void k_conv_wT(const float* __restrict__ in,
                          uint16_t* __restrict__ out) {
    int i = blockIdx.x * blockDim.x + threadIdx.x;
    int k = i >> 10, n = i & 1023;
    out[n * 1024 + k] = f2bf(in[i]);
}

// ---------------------------------------------------------------------------
// RoPE (half-split) on [B,H,T,D] f32 -> bf16, scale folded in (1/128 on q)
__global__ void k_rope(const float* __restrict__ in,
                       uint16_t* __restrict__ out, float scale) {
    int i = blockIdx.x * blockDim.x + threadIdx.x;   // over B*H*T*64
    int d   = i & 63;
    int t   = (i >> 6) & (T_ - 1);
    int bht = i >> 18;                               // b*H+h (0..15)
    if (bht >= B_ * H_) return;
    int base = (bht * T_ + t) * D_;
    float e = in[base + d];
    float o = in[base + d + 64];
    float freq = powf(10000.0f, -(float)d / 64.0f);
    float rad = (float)t * freq;
    float s = sinf(rad), c = cosf(rad);
    out[base + d]      = f2bf((e * c - o * s) * scale);
    out[base + d + 64] = f2bf((e * s + o * c) * scale);
}

// ---------------------------------------------------------------------------
// bf16 WMMA GEMM: C[8192 x 1024] = A[8192 x 1024] * B, B stored transposed [1024 x 1024].
// mode 0: f32 out row-major; mode 1: f32 -> [B,H,T,D]; mode 2: bf16 -> [B,H,D,T].
// Block = 256 thr (8 waves); block tile 256 x 64; wave tile 32 x 64.
__global__ void k_gemm(const uint16_t* __restrict__ Abf,
                       const uint16_t* __restrict__ BT,
                       float* __restrict__ outF,
                       uint16_t* __restrict__ outBf,
                       int mode) {
    constexpr int K = 1024, N = 1024;
    int w = threadIdx.x >> 5, lane = threadIdx.x & 31;
    int hi = lane >> 4, l16 = lane & 15;
    int r0 = blockIdx.x * 256 + w * 32;
    int n0 = blockIdx.y * 64;

    Acc acc[2][4];
#pragma unroll
    for (int rb = 0; rb < 2; ++rb)
#pragma unroll
        for (int nt = 0; nt < 4; ++nt)
#pragma unroll
            for (int r = 0; r < 8; ++r) acc[rb][nt].f[r] = 0.0f;

    const char* abase = (const char*)Abf + (size_t)(r0 + l16) * K * 2;
    const char* bbase = (const char*)BT + (size_t)(n0 + l16) * K * 2;

    for (int k0 = 0; k0 < K; k0 += 32) {
        Frag a0 = load_afrag(abase + k0 * 2, hi);
        Frag a1 = load_afrag(abase + k0 * 2 + 16 * K * 2, hi);
#pragma unroll
        for (int nt = 0; nt < 4; ++nt) {
            Frag b = load_bfrag(bbase + nt * (16 * K * 2) + k0 * 2, hi);
            acc[0][nt].v = wmma_bf16(a0.v, b.v, acc[0][nt].v);
            acc[1][nt].v = wmma_bf16(a1.v, b.v, acc[1][nt].v);
        }
    }

#pragma unroll
    for (int rb = 0; rb < 2; ++rb)
#pragma unroll
        for (int nt = 0; nt < 4; ++nt)
#pragma unroll
            for (int r = 0; r < 8; ++r) {
                int row = r0 + rb * 16 + r + hi * 8;
                int col = n0 + nt * 16 + l16;
                float v = acc[rb][nt].f[r];
                if (mode == 0) {
                    outF[(size_t)row * N + col] = v;
                } else {
                    int b = row >> 12, t = row & (T_ - 1);
                    int h = col >> 7, d = col & (D_ - 1);
                    if (mode == 1)
                        outF[((size_t)(b * H_ + h) * T_ + t) * D_ + d] = v;
                    else
                        outBf[((size_t)(b * H_ + h) * D_ + d) * T_ + t] = f2bf(v);
                }
            }
}

// ---------------------------------------------------------------------------
// Flash attention: q,k bf16 [B,H,T,D] (q pre-scaled by 1/128), v bf16 [B,H,D,T].
// Out: o bf16 [B*T, H*D]. Block = 256 thr = 8 waves; wave = 16 q rows, 64-key blocks.
__global__ void k_attn(const uint16_t* __restrict__ qbf,
                       const uint16_t* __restrict__ kbf,
                       const uint16_t* __restrict__ vbf,
                       uint16_t* __restrict__ obf) {
    __shared__ uint16_t plds[8][16][64];
    int w = threadIdx.x >> 5, lane = threadIdx.x & 31;
    int hi = lane >> 4, l16 = lane & 15;
    int bh = blockIdx.y;              // b*H + h
    int b = bh >> 3, h = bh & 7;
    int q0 = blockIdx.x * 128 + w * 16;

    // Resident q A-frags (4 chunks of 32 along D)
    const char* qbase = (const char*)qbf + ((size_t)(bh * T_ + q0 + l16) * D_) * 2;
    Frag qf[4];
#pragma unroll
    for (int dc = 0; dc < 4; ++dc) qf[dc] = load_afrag(qbase + dc * 64, hi);

    // Hoisted key/value base pointers (per-lane)
    const char* kbase = (const char*)kbf + ((size_t)(bh * T_ + l16) * D_) * 2;
    const char* vbase[8];
#pragma unroll
    for (int dt = 0; dt < 8; ++dt)
        vbase[dt] = (const char*)vbf + ((size_t)(bh * D_ + dt * 16 + l16) * T_) * 2;

    char* plbase = (char*)&plds[w][0][0] + l16 * 128;

    Acc oacc[8];
    float m[8], l[8];
#pragma unroll
    for (int dt = 0; dt < 8; ++dt)
#pragma unroll
        for (int r = 0; r < 8; ++r) oacc[dt].f[r] = 0.0f;
#pragma unroll
    for (int r = 0; r < 8; ++r) { m[r] = -1e30f; l[r] = 0.0f; }

    const int lastRow = q0 + 15;
    for (int j0 = 0; j0 <= lastRow; j0 += 64) {
        const char* kp = kbase + j0 * (D_ * 2);
        // Prefetch next key block (speculative; dropped if invalid)
        __builtin_prefetch(kp + 64 * (D_ * 2), 0, 1);

        // S = q * k^T for 64 keys (4 N-tiles of 16)
        Acc sacc[4];
#pragma unroll
        for (int jt = 0; jt < 4; ++jt) {
#pragma unroll
            for (int r = 0; r < 8; ++r) sacc[jt].f[r] = 0.0f;
#pragma unroll
            for (int dc = 0; dc < 4; ++dc) {
                Frag kf = load_bfrag(kp + jt * (16 * D_ * 2) + dc * 64, hi);
                sacc[jt].v = wmma_bf16(qf[dc].v, kf.v, sacc[jt].v);
            }
        }

        // Causal mask + online softmax (rows live in 16-lane halves)
        float scl[8];
#pragma unroll
        for (int r = 0; r < 8; ++r) {
            int rowg = q0 + r + hi * 8;
            float mx = -1e30f;
#pragma unroll
            for (int jt = 0; jt < 4; ++jt) {
                int j = j0 + jt * 16 + l16;
                float sv = sacc[jt].f[r];
                if (j > rowg) { sv = -1e30f; sacc[jt].f[r] = sv; }
                mx = fmaxf(mx, sv);
            }
#pragma unroll
            for (int off = 1; off < 16; off <<= 1)
                mx = fmaxf(mx, __shfl_xor(mx, off, 32));
            float mnew = fmaxf(m[r], mx);
            float sc = __expf(m[r] - mnew);
            m[r] = mnew;
            float ps = 0.0f;
#pragma unroll
            for (int jt = 0; jt < 4; ++jt) {
                float p = __expf(sacc[jt].f[r] - mnew);
                sacc[jt].f[r] = p;
                ps += p;
            }
#pragma unroll
            for (int off = 1; off < 16; off <<= 1)
                ps += __shfl_xor(ps, off, 32);
            l[r] = l[r] * sc + ps;
            scl[r] = sc;
        }
#pragma unroll
        for (int dt = 0; dt < 8; ++dt)
#pragma unroll
            for (int r = 0; r < 8; ++r) oacc[dt].f[r] *= scl[r];

        // P -> per-wave LDS tile, re-layout into A-frags
#pragma unroll
        for (int jt = 0; jt < 4; ++jt)
#pragma unroll
            for (int r = 0; r < 8; ++r)
                plds[w][r + hi * 8][jt * 16 + l16] = f2bf(sacc[jt].f[r]);
        asm volatile("s_wait_dscnt 0" ::: "memory");

        Frag pf[2];
#pragma unroll
        for (int c = 0; c < 2; ++c) {
            pf[c].q[0] = *(const uint4*)(plbase + c * 64 + hi * 16);
            pf[c].q[1] = *(const uint4*)(plbase + c * 64 + hi * 16 + 32);
        }

        // O += P * V  (V transposed [B,H,D,T] -> contiguous b128 B-frags)
#pragma unroll
        for (int dt = 0; dt < 8; ++dt) {
            const char* vp = vbase[dt] + j0 * 2;
#pragma unroll
            for (int c = 0; c < 2; ++c) {
                Frag vf = load_bfrag(vp + c * 64, hi);
                oacc[dt].v = wmma_bf16(pf[c].v, vf.v, oacc[dt].v);
            }
        }
        asm volatile("s_wait_dscnt 0" ::: "memory");
    }

    // Finalize: divide by l, store bf16 to [B*T, H*D]
#pragma unroll
    for (int dt = 0; dt < 8; ++dt)
#pragma unroll
        for (int r = 0; r < 8; ++r) {
            int rowg = q0 + r + hi * 8;
            int col = h * D_ + dt * 16 + l16;
            float v = oacc[dt].f[r] / l[r];
            obf[((size_t)b * T_ + rowg) * HD_ + col] = f2bf(v);
        }
}

// ---------------------------------------------------------------------------
extern "C" void kernel_launch(void* const* d_in, const int* in_sizes, int n_in,
                              void* d_out, int out_size, void* d_ws, size_t ws_size,
                              hipStream_t stream) {
    (void)in_sizes; (void)n_in; (void)out_size; (void)ws_size;
    const float* x    = (const float*)d_in[0];
    const float* w_aq = (const float*)d_in[1];
    const float* w_ak = (const float*)d_in[2];
    const float* w_av = (const float*)d_in[3];
    const float* w_ao = (const float*)d_in[4];
    float* out = (float*)d_out;

    char* ws = (char*)d_ws;
    const size_t MB = 1024 * 1024;
    uint16_t* x_bf = (uint16_t*)(ws + 0);          // 16 MB  [B*T, M] bf16
    uint16_t* wqT  = (uint16_t*)(ws + 16 * MB);    //  2 MB  [N,K] bf16
    uint16_t* wkT  = (uint16_t*)(ws + 18 * MB);
    uint16_t* wvT  = (uint16_t*)(ws + 20 * MB);
    uint16_t* woT  = (uint16_t*)(ws + 22 * MB);
    float*    q32  = (float*)   (ws + 24 * MB);    // 32 MB  [B,H,T,D] f32
    float*    k32  = (float*)   (ws + 56 * MB);    // 32 MB
    uint16_t* q_bf = (uint16_t*)(ws + 88 * MB);    // 16 MB  [B,H,T,D] bf16
    uint16_t* k_bf = (uint16_t*)(ws + 104 * MB);   // 16 MB
    uint16_t* v_bf = (uint16_t*)(ws + 120 * MB);   // 16 MB  [B,H,D,T] bf16
    uint16_t* o_bf = (uint16_t*)(ws + 136 * MB);   // 16 MB  [B*T, H*D] bf16

    const int nX = BT_ * M_;       // 8388608
    const int nW = M_ * HD_;       // 1048576

    k_conv_x4<<<nX / 4 / 256, 256, 0, stream>>>((const float4*)x, (ushort4*)x_bf, nX / 4);
    k_conv_wT<<<nW / 256, 256, 0, stream>>>(w_aq, wqT);
    k_conv_wT<<<nW / 256, 256, 0, stream>>>(w_ak, wkT);
    k_conv_wT<<<nW / 256, 256, 0, stream>>>(w_av, wvT);
    k_conv_wT<<<nW / 256, 256, 0, stream>>>(w_ao, woT);

    dim3 gg(BT_ / 256, HD_ / 64);  // 32 x 16
    k_gemm<<<gg, 256, 0, stream>>>(x_bf, wqT, q32, nullptr, 1);
    k_gemm<<<gg, 256, 0, stream>>>(x_bf, wkT, k32, nullptr, 1);
    k_gemm<<<gg, 256, 0, stream>>>(x_bf, wvT, nullptr, v_bf, 2);

    const int nR = B_ * H_ * T_ * 64;  // 4194304
    k_rope<<<nR / 256, 256, 0, stream>>>(q32, q_bf, 1.0f / 128.0f);
    k_rope<<<nR / 256, 256, 0, stream>>>(k32, k_bf, 1.0f);

    dim3 ga(T_ / 128, B_ * H_);    // 32 x 16
    k_attn<<<ga, 256, 0, stream>>>(q_bf, k_bf, v_bf, o_bf);

    k_gemm<<<gg, 256, 0, stream>>>(o_bf, woT, out, nullptr, 0);
}
